// CustomMultiheadAttention_39384850104840
// MI455X (gfx1250) — matrix operand order repaired and checked by reading, not compiled, not run on hardware
//
#include <hip/hip_runtime.h>
#include <hip/hip_bf16.h>

// ---------------- types ----------------
typedef __attribute__((ext_vector_type(16))) __bf16 v16bf;
typedef __attribute__((ext_vector_type(8)))  float  v8f;
typedef __attribute__((ext_vector_type(4)))  int    v4i;

union FragBF { v16bf v; uint4 q[2]; };

// epilogue modes
constexpr int EP_F32       = 0;  // store fp32 (+bias)
constexpr int EP_BF16      = 1;  // store bf16 (+bias)
constexpr int EP_SCALE_F32 = 2;  // store fp32 * scale
constexpr int EP_GELU_BF16 = 3;  // store bf16(gelu(x+bias)), exact erf

// ---------------- helpers ----------------
__device__ __forceinline__ unsigned short f2bf(float f) {
  union { float f; unsigned u; } v; v.f = f;
  unsigned u = v.u;
  unsigned r = u + 0x7FFFu + ((u >> 16) & 1u);   // round-to-nearest-even
  return (unsigned short)(r >> 16);
}

// load 4 consecutive elements, return 4 packed bf16 (2 dwords)
template <typename T> __device__ __forceinline__ uint2 load4_bf16(const T* p);

template <> __device__ __forceinline__ uint2 load4_bf16<float>(const float* p) {
  float4 f = *reinterpret_cast<const float4*>(p);
  uint2 r;
  r.x = (unsigned)f2bf(f.x) | ((unsigned)f2bf(f.y) << 16);
  r.y = (unsigned)f2bf(f.z) | ((unsigned)f2bf(f.w) << 16);
  return r;
}
template <> __device__ __forceinline__ uint2 load4_bf16<unsigned short>(const unsigned short* p) {
  return *reinterpret_cast<const uint2*>(p);  // already bf16
}

// ---------------- CDNA5 async global->LDS copy (ASYNCcnt-tracked) ----------------
#if __has_builtin(__builtin_amdgcn_global_load_async_to_lds_b128)
#define HAVE_ASYNC_B128 1
__device__ __forceinline__ void async_copy_b128(void* lds, const void* g) {
  __builtin_amdgcn_global_load_async_to_lds_b128(
      (__attribute__((address_space(1))) v4i*)g,
      (__attribute__((address_space(3))) v4i*)lds, 0, 0);
}
__device__ __forceinline__ void async_wait0() {
#if __has_builtin(__builtin_amdgcn_s_wait_asynccnt)
  __builtin_amdgcn_s_wait_asynccnt(0);
#else
  asm volatile("s_wait_asynccnt 0x0" ::: "memory");
#endif
}
#else
#define HAVE_ASYNC_B128 0
__device__ __forceinline__ void async_copy_b128(void*, const void*) {}
__device__ __forceinline__ void async_wait0() {}
#endif

// LDS row stride (in bf16 elems): 32 K + 8 pad -> 80B rows (16B-aligned), no bank conflicts
#define LDS_STR 40

// A fragment: 16x32 bf16. lane&15 = M row; lane>>4 selects K-halves per ISA:
//   elems 0..7  -> K = i + khalf*8          (VGPR0..3)
//   elems 8..15 -> K = 16 + (i-8) + khalf*8 (VGPR4..7)
__device__ __forceinline__ v16bf load_frag_a(const unsigned short* As, int m_base, int lane) {
  const unsigned short* row = As + (m_base + (lane & 15)) * LDS_STR;
  const int kh = (lane >> 4) * 8;
  FragBF f;
  f.q[0] = *reinterpret_cast<const uint4*>(row + kh);
  f.q[1] = *reinterpret_cast<const uint4*>(row + 16 + kh);
  return f.v;
}

// B fragment: 32x16 bf16. lane&15 = N col; K order matches A: elems 0..7 -> K = i + khalf*8, elems 8..15 -> K = 16 + (i-8) + khalf*8.
// Bs stored [n][k] so this is two 16-byte chunks per lane.
__device__ __forceinline__ v16bf load_frag_b(const unsigned short* Bs, int n_base, int lane) {
  const unsigned short* row = Bs + (n_base + (lane & 15)) * LDS_STR + (lane >> 4) * 8;
  FragBF f;
  f.q[0] = *reinterpret_cast<const uint4*>(row);
  f.q[1] = *reinterpret_cast<const uint4*>(row + 16);
  return f.v;
}

// ---------------- GEMM: C[M,N] = op(A[M,K] @ B) ----------------
// BT=false: B is row-major [K,N] (ldb = row stride).
// BT=true : B source is row-major [N,K] and we compute A @ B^T (ldb = K-row stride).
// Block = 128x128, 256 threads (8 waves as 4Mx2N), wave tile 32x64 = 2x4 WMMA tiles.
template <typename TA, typename TB, int MODE, bool BT, bool HAS_BIAS>
__global__ __launch_bounds__(256) void gemm_wmma(
    const TA* __restrict__ A, const TB* __restrict__ B, void* __restrict__ Cv,
    const float* __restrict__ bias, int M, int N, int K,
    int lda, int ldb, int ldc, long long sA, long long sB, long long sC, float scale)
{
  __shared__ __align__(16) unsigned short As[128 * LDS_STR];
  __shared__ __align__(16) unsigned short Bs[128 * LDS_STR];

  constexpr bool ASYNC_A = (sizeof(TA) == 2) && HAVE_ASYNC_B128;
  constexpr bool ASYNC_B = BT && (sizeof(TB) == 2) && HAVE_ASYNC_B128;

  const int bz = blockIdx.z;
  A += (long long)bz * sA;
  B += (long long)bz * sB;

  const int n0 = blockIdx.x * 128;
  const int m0 = blockIdx.y * 128;
  const int t    = threadIdx.x;
  const int lane = t & 31;
  const int wave = t >> 5;
  const int wm = (wave & 3) * 32;   // wave M offset in block
  const int wn = (wave >> 2) * 64;  // wave N offset in block

  v8f acc[2][4] = {};

  for (int k0 = 0; k0 < K; k0 += 32) {
    // ---- stage A tile 128x32 bf16 ----
    if constexpr (ASYNC_A) {
      // raw bf16 bytes: DMA straight into LDS, 16B per lane, 512 chunks / 256 threads
#pragma unroll
      for (int r = 0; r < 2; ++r) {
        int id = t + r * 256;            // 0..511
        int m  = id >> 2;
        int k  = (id & 3) * 8;
        async_copy_b128(&As[m * LDS_STR + k],
                        A + (long long)(m0 + m) * lda + (k0 + k));
      }
    } else {
#pragma unroll
      for (int r = 0; r < 4; ++r) {
        int id = t + r * 256;            // 0..1023 float4-groups
        int m  = id >> 3;
        int k  = (id & 7) * 4;
        uint2 p = load4_bf16(A + (long long)(m0 + m) * lda + (k0 + k));
        *reinterpret_cast<uint2*>(&As[m * LDS_STR + k]) = p;
      }
    }
    // ---- stage B tile 32x128 into Bs[n][k] ----
    if constexpr (ASYNC_B) {
#pragma unroll
      for (int r = 0; r < 2; ++r) {
        int id = t + r * 256;
        int n  = id >> 2;
        int k  = (id & 3) * 8;
        async_copy_b128(&Bs[n * LDS_STR + k],
                        B + (long long)(n0 + n) * ldb + (k0 + k));
      }
    } else if constexpr (BT) {
#pragma unroll
      for (int r = 0; r < 4; ++r) {
        int id = t + r * 256;
        int n  = id >> 3;
        int k  = (id & 7) * 4;
        uint2 p = load4_bf16(B + (long long)(n0 + n) * ldb + (k0 + k));
        *reinterpret_cast<uint2*>(&Bs[n * LDS_STR + k]) = p;
      }
    } else {
#pragma unroll
      for (int r = 0; r < 4; ++r) {
        int id = t + r * 256;
        int k  = id >> 5;
        int n  = (id & 31) * 4;
        uint2 p = load4_bf16(B + (long long)(k0 + k) * ldb + (n0 + n));
        Bs[(n + 0) * LDS_STR + k] = (unsigned short)(p.x);
        Bs[(n + 1) * LDS_STR + k] = (unsigned short)(p.x >> 16);
        Bs[(n + 2) * LDS_STR + k] = (unsigned short)(p.y);
        Bs[(n + 3) * LDS_STR + k] = (unsigned short)(p.y >> 16);
      }
    }
    if constexpr (ASYNC_A || ASYNC_B) async_wait0();
    __syncthreads();

    v16bf afrag[2];
#pragma unroll
    for (int mi = 0; mi < 2; ++mi) afrag[mi] = load_frag_a(As, wm + mi * 16, lane);
#pragma unroll
    for (int ni = 0; ni < 4; ++ni) {
      v16bf bfrag = load_frag_b(Bs, wn + ni * 16, lane);
#pragma unroll
      for (int mi = 0; mi < 2; ++mi) {
        acc[mi][ni] = __builtin_amdgcn_wmma_f32_16x16x32_bf16(
            false, afrag[mi], false, bfrag, (short)0, acc[mi][ni], false, false);
      }
    }
    __syncthreads();
  }

  // ---- epilogue: C/D layout: lane&15 = N, VGPR v -> M = v + 8*(lane>=16) ----
  const int nlane = lane & 15;
  const int rhalf = (lane >> 4) * 8;
#pragma unroll
  for (int mi = 0; mi < 2; ++mi) {
#pragma unroll
    for (int ni = 0; ni < 4; ++ni) {
      const int col = n0 + wn + ni * 16 + nlane;
      float bval = 0.0f;
      if constexpr (HAS_BIAS) bval = bias[col];
#pragma unroll
      for (int v = 0; v < 8; ++v) {
        const int row = m0 + wm + mi * 16 + rhalf + v;
        const long long off = (long long)bz * sC + (long long)row * ldc + col;
        float x = acc[mi][ni][v] + bval;
        if (MODE == EP_F32) {
          ((float*)Cv)[off] = x;
        } else if (MODE == EP_SCALE_F32) {
          ((float*)Cv)[off] = x * scale;
        } else if (MODE == EP_BF16) {
          ((unsigned short*)Cv)[off] = f2bf(x);
        } else {  // EP_GELU_BF16, exact gelu
          float g = 0.5f * x * (1.0f + erff(x * 0.70710678118654752f));
          ((unsigned short*)Cv)[off] = f2bf(g);
        }
      }
    }
  }
}

// ---------------- wave/block reductions (wave32) ----------------
__device__ __forceinline__ float wave_max(float v) {
#pragma unroll
  for (int i = 16; i >= 1; i >>= 1) v = fmaxf(v, __shfl_xor(v, i, 32));
  return v;
}
__device__ __forceinline__ float wave_sum(float v) {
#pragma unroll
  for (int i = 16; i >= 1; i >>= 1) v += __shfl_xor(v, i, 32);
  return v;
}

// ---------------- softmax in place over rows of 1024 fp32 ----------------
__global__ __launch_bounds__(256) void softmax_kernel(float* __restrict__ S) {
  __shared__ float redm[8];
  __shared__ float reds[8];
  const int t = threadIdx.x, lane = t & 31, wave = t >> 5;
  float* row = S + (long long)blockIdx.x * 1024;
  float4 x = *reinterpret_cast<float4*>(row + t * 4);
  float m = fmaxf(fmaxf(x.x, x.y), fmaxf(x.z, x.w));
  m = wave_max(m);
  if (lane == 0) redm[wave] = m;
  __syncthreads();
  float bm = redm[0];
#pragma unroll
  for (int i = 1; i < 8; ++i) bm = fmaxf(bm, redm[i]);
  float e0 = expf(x.x - bm), e1 = expf(x.y - bm);
  float e2 = expf(x.z - bm), e3 = expf(x.w - bm);
  float s = wave_sum(e0 + e1 + e2 + e3);
  if (lane == 0) reds[wave] = s;
  __syncthreads();
  float bs = reds[0];
#pragma unroll
  for (int i = 1; i < 8; ++i) bs += reds[i];
  const float inv = 1.0f / bs;
  float4 y; y.x = e0 * inv; y.y = e1 * inv; y.z = e2 * inv; y.w = e3 * inv;
  *reinterpret_cast<float4*>(row + t * 4) = y;
}

// ---------------- layernorm rows of 512 fp32 -> bf16 ----------------
__global__ __launch_bounds__(256) void layernorm_kernel(
    const float* __restrict__ X, const float* __restrict__ gamma,
    const float* __restrict__ beta, unsigned short* __restrict__ Y)
{
  __shared__ float rs[8], rq[8];
  const int t = threadIdx.x, lane = t & 31, wave = t >> 5;
  const float* row = X + (long long)blockIdx.x * 512;
  float2 x = *reinterpret_cast<const float2*>(row + t * 2);
  float s = wave_sum(x.x + x.y);
  float q = wave_sum(x.x * x.x + x.y * x.y);
  if (lane == 0) { rs[wave] = s; rq[wave] = q; }
  __syncthreads();
  float S = 0.0f, Q = 0.0f;
#pragma unroll
  for (int i = 0; i < 8; ++i) { S += rs[i]; Q += rq[i]; }
  const float mean = S * (1.0f / 512.0f);
  const float var  = Q * (1.0f / 512.0f) - mean * mean;
  const float rstd = rsqrtf(var + 1e-5f);
  float y0 = (x.x - mean) * rstd * gamma[t * 2]     + beta[t * 2];
  float y1 = (x.y - mean) * rstd * gamma[t * 2 + 1] + beta[t * 2 + 1];
  unsigned packed = (unsigned)f2bf(y0) | ((unsigned)f2bf(y1) << 16);
  *reinterpret_cast<unsigned*>(Y + (long long)blockIdx.x * 512 + t * 2) = packed;
}

// ---------------- launcher ----------------
extern "C" void kernel_launch(void* const* d_in, const int* in_sizes, int n_in,
                              void* d_out, int out_size, void* d_ws, size_t ws_size,
                              hipStream_t stream) {
  (void)in_sizes; (void)n_in; (void)out_size; (void)ws_size;
  const float* query = (const float*)d_in[0];
  const float* key_  = (const float*)d_in[1];
  const float* value = (const float*)d_in[2];
  const float* Wq = (const float*)d_in[3];
  const float* bq = (const float*)d_in[4];
  const float* Wk = (const float*)d_in[5];
  const float* bk = (const float*)d_in[6];
  const float* Wv = (const float*)d_in[7];
  const float* bv = (const float*)d_in[8];
  const float* ln_g = (const float*)d_in[9];
  const float* ln_b = (const float*)d_in[10];
  const float* W1 = (const float*)d_in[11];
  const float* b1 = (const float*)d_in[12];
  const float* W2 = (const float*)d_in[13];
  const float* b2 = (const float*)d_in[14];
  const float* W3 = (const float*)d_in[15];
  const float* b3 = (const float*)d_in[16];

  float* out  = (float*)d_out;                       // [32,1024,512]
  float* attn = out + (long long)32768 * 512;        // [32,1024,1024]

  char* ws = (char*)d_ws;
  unsigned short* Qb = (unsigned short*)(ws);                 // 32768x512 bf16
  unsigned short* Kb = (unsigned short*)(ws + 33554432);      // 32768x512 bf16
  unsigned short* Vb = (unsigned short*)(ws + 67108864);      // 32768x512 bf16
  float* attn_out    = (float*)(ws + 100663296);              // 32768x512 f32
  unsigned short* h1b = (unsigned short*)(ws + 167772160);    // 32768x2048 bf16
  unsigned short* h2b = (unsigned short*)(ws + 301989888);    // 32768x2048 bf16
  unsigned short* hb  = Qb;  // LN output reuses dead Q buffer

  const float scale = 0.044194173824159216f;  // 1/sqrt(512)
  dim3 blk(256);

  // QKV projections: [32768,512] = X[32768,512] @ W[512,512] (+b), -> bf16
  gemm_wmma<float, float, EP_BF16, false, true><<<dim3(4, 256, 1), blk, 0, stream>>>(
      query, Wq, Qb, bq, 32768, 512, 512, 512, 512, 512, 0, 0, 0, 1.0f);
  gemm_wmma<float, float, EP_BF16, false, true><<<dim3(4, 256, 1), blk, 0, stream>>>(
      key_, Wk, Kb, bk, 32768, 512, 512, 512, 512, 512, 0, 0, 0, 1.0f);
  gemm_wmma<float, float, EP_BF16, false, true><<<dim3(4, 256, 1), blk, 0, stream>>>(
      value, Wv, Vb, bv, 32768, 512, 512, 512, 512, 512, 0, 0, 0, 1.0f);

  // scores[b] = Q[b] @ K[b]^T * scale -> fp32, written straight into d_out attn region
  gemm_wmma<unsigned short, unsigned short, EP_SCALE_F32, true, false>
      <<<dim3(8, 8, 32), blk, 0, stream>>>(
      Qb, Kb, attn, nullptr, 1024, 1024, 512, 512, 512, 1024,
      524288, 524288, 1048576, scale);

  // softmax in place (attn region is the 2nd reference output)
  softmax_kernel<<<32768, blk, 0, stream>>>(attn);

  // attn_out[b] = attn[b] @ V[b]  (fp32 A converted on the fly)
  gemm_wmma<float, unsigned short, EP_F32, false, false><<<dim3(4, 8, 32), blk, 0, stream>>>(
      attn, Vb, attn_out, nullptr, 1024, 512, 1024, 1024, 512, 512,
      1048576, 524288, 524288, 1.0f);

  // layernorm -> bf16
  layernorm_kernel<<<32768, blk, 0, stream>>>(attn_out, ln_g, ln_b, hb);

  // FFN: gelu(h @ W1 + b1) -> gelu(. @ W2 + b2) -> (. @ W3 + b3)
  gemm_wmma<unsigned short, float, EP_GELU_BF16, false, true><<<dim3(16, 256, 1), blk, 0, stream>>>(
      hb, W1, h1b, b1, 32768, 2048, 512, 512, 2048, 2048, 0, 0, 0, 1.0f);
  gemm_wmma<unsigned short, float, EP_GELU_BF16, false, true><<<dim3(16, 256, 1), blk, 0, stream>>>(
      h1b, W2, h2b, b2, 32768, 2048, 2048, 2048, 2048, 2048, 0, 0, 0, 1.0f);
  gemm_wmma<unsigned short, float, EP_F32, false, true><<<dim3(4, 256, 1), blk, 0, stream>>>(
      h2b, W3, out, b3, 32768, 512, 2048, 2048, 512, 512, 0, 0, 0, 1.0f);
}
